// StructureConsensuLossFunction_429496730161
// MI455X (gfx1250) — matrix operand order repaired and checked by reading, not compiled, not run on hardware
//
#include <hip/hip_runtime.h>
#include <hip/hip_bf16.h>
#include <climits>

// ---------------------------------------------------------------------------
// StructureConsensusLoss on gfx1250 (MI455X)
// Phase 1: stream logits once (HBM-bound ~8us), per-pixel softmax,
//          segmented sums via one-hot f16 WMMA (v_wmma_f32_16x16x32_f16).
//          LDS staging is [channel][pixel] so WMMA B operands load as
//          contiguous ds_load_b128, not per-element u16 gathers.
// Phase 2: tiny finalize kernel -> scalar.
// ---------------------------------------------------------------------------

typedef __attribute__((ext_vector_type(16))) _Float16 v16h;
typedef __attribute__((ext_vector_type(8)))  _Float16 v8h;
typedef __attribute__((ext_vector_type(8)))  float    v8f;

namespace scl {
constexpr int N = 8, C = 21, H = 512, W = 512, S = 16;
constexpr int P = H * W;                 // 262144 pixels per image
constexpr int CP = 32;                   // channels padded to 32 (2 WMMA N-tiles)
constexpr int BLK_PER_IMG = 64;
constexpr int WAVES_PER_BLK = 8;         // 256 threads, wave32
constexpr int TILES_PER_IMG = P / 32;    // 8192 (exactly divisible)
constexpr int WAVE_STRIDE = BLK_PER_IMG * WAVES_PER_BLK;  // 512 waves/image
constexpr int KSTR = 40;                 // staging row stride in halves (80B)
constexpr float ALPHA = 10.0f, BETA = 5.0f;
constexpr float L2E = 1.4426950408889634f;   // log2(e)
constexpr float LN2 = 0.6931471805599453f;   // ln(2)
}

static __device__ __forceinline__ v16h join8(v8h lo, v8h hi) {
  return __builtin_shufflevector(lo, hi, 0, 1, 2, 3, 4, 5, 6, 7, 8, 9, 10, 11,
                                 12, 13, 14, 15);
}

// ---------------------------------------------------------------------------
// Workspace zero/init: [gsum: N*S*CP f32][gx: N*S*CP f32][labels: S i32]
// ---------------------------------------------------------------------------
__global__ void scl_zero(float* ws_f, int nfloats, int* labels, int nlab) {
  int i = blockIdx.x * blockDim.x + threadIdx.x;
  if (i < nfloats) ws_f[i] = 0.0f;
  if (i < nlab)    labels[i] = INT_MIN;
}

// ---------------------------------------------------------------------------
// Phase 1: softmax + segmented accumulation through WMMA one-hot GEMM.
// ---------------------------------------------------------------------------
__global__ __launch_bounds__(256) void scl_phase1(
    const float* __restrict__ logit, const int* __restrict__ blobs,
    const int* __restrict__ target, float* __restrict__ gsum,
    float* __restrict__ gx, int* __restrict__ glab) {
  using namespace scl;
  // [channel 0..31][pixel 0..31], row stride 40 halves: ds_load_b128-friendly.
  // stX has one extra row (32) holding sid as f16 for the A-matrix build.
  __shared__ alignas(16) _Float16 stP[WAVES_PER_BLK][32][KSTR];
  __shared__ alignas(16) _Float16 stX[WAVES_PER_BLK][33][KSTR];
  __shared__ int labs[S];

  const int tid  = threadIdx.x;
  const int lane = tid & 31;
  const int wv   = tid >> 5;
  const int hlf  = lane >> 4;     // 0: K-lo half, 1: K-hi half
  const int m    = lane & 15;     // A row (segment) / B,D column
  const _Float16 mh = (_Float16)(float)m;

  if (tid < S) labs[tid] = INT_MIN;
  __syncthreads();

  const int n        = blockIdx.x / BLK_PER_IMG;
  const int blkInImg = blockIdx.x % BLK_PER_IMG;
  const int widx     = blkInImg * WAVES_PER_BLK + wv;

  const float* lg = logit + (size_t)n * C * P;
  const int*   bl = blobs + (size_t)n * P;
  const int*   tg = target + (size_t)n * P;

  _Float16 (*sp)[KSTR] = stP[wv];
  _Float16 (*sx)[KSTR] = stX[wv];

  // Constant staging rows, written once (wave-private; LDS is in-order/wave):
  // p row 21 = 1.0 (exact per-segment count column), rows 22..31 = 0 pad.
  sp[21][lane] = (_Float16)1.0f;
#pragma unroll
  for (int r = 22; r < 32; ++r) {
    sp[r][lane] = (_Float16)0.0f;
    sx[r][lane] = (_Float16)0.0f;
  }

  v8f dp0 = {}, dp1 = {}, dx0 = {}, dx1 = {};

  // Uniform trip count (TILES_PER_IMG % WAVE_STRIDE == 0): EXEC stays all-1s.
  // Loop body is branchless (prefetch is speculative -> safe past the end).
  for (int t = widx; t < TILES_PER_IMG; t += WAVE_STRIDE) {
    const int pix = t * 32 + lane;
    const int sid = bl[pix];                        // 0..15
    atomicMax(&labs[sid], tg[pix]);                 // faithful segment_max

    __builtin_prefetch(&bl[pix + WAVE_STRIDE * 32], 0, 3);
    __builtin_prefetch(&lg[pix + WAVE_STRIDE * 32], 0, 3);

    // 21 coalesced channel loads (channel stride = P).
    float x[C];
#pragma unroll
    for (int c = 0; c < C; ++c) x[c] = lg[(size_t)c * P + pix];

    float mx = x[0];
#pragma unroll
    for (int c = 1; c < C; ++c) mx = fmaxf(mx, x[c]);
    const float nb = -mx * L2E;                     // exp(x-mx)=exp2(x*L2E+nb)

    // Keep prior iteration's LDS reads (program-order earlier) ahead of the
    // overwrite below; hardware keeps same-wave LDS ops in order.
    __builtin_amdgcn_wave_barrier();

    float sum = 0.0f;
#pragma unroll
    for (int c = 0; c < C; ++c) {
      sx[c][lane] = (_Float16)x[c];                 // stage raw logits
      float e = __builtin_amdgcn_exp2f(fmaf(x[c], L2E, nb));
      x[c] = e;                                     // x[] now holds exp
      sum += e;
    }
    const float inv = 1.0f / sum;
    const float off = fmaf(LN2, __builtin_amdgcn_logf(sum), mx);  // log p=x-off
    sx[21][lane] = (_Float16)off;                   // per-pixel correction row
    sx[32][lane] = (_Float16)(float)sid;            // sid row for A build
#pragma unroll
    for (int c = 0; c < C; ++c) sp[c][lane] = (_Float16)(x[c] * inv);  // p

    __builtin_amdgcn_wave_barrier();

    // A = one-hot(segment) 16x32 f16. Element i maps to
    // K = 16*hlf + i: one contiguous 16-half run.
    const v8h q0 = *(const v8h*)&sx[32][16 * hlf];
    const v8h q1 = *(const v8h*)&sx[32][16 * hlf + 8];
    v16h a;
#pragma unroll
    for (int i = 0; i < 8; ++i) {
      a[i]     = (q0[i] == mh) ? (_Float16)1.0f : (_Float16)0.0f;
      a[i + 8] = (q1[i] == mh) ? (_Float16)1.0f : (_Float16)0.0f;
    }

    // B tiles: lane's 16 halves = K = 16*hlf + 0..15 at fixed column
    // (= staging row). Contiguous in LDS -> two b128 loads per operand.
    const int kb = 16 * hlf;
    const v16h bp0 = join8(*(const v8h*)&sp[m][kb],      *(const v8h*)&sp[m][kb + 8]);
    const v16h bp1 = join8(*(const v8h*)&sp[16 + m][kb], *(const v8h*)&sp[16 + m][kb + 8]);
    const v16h bx0 = join8(*(const v8h*)&sx[m][kb],      *(const v8h*)&sx[m][kb + 8]);
    const v16h bx1 = join8(*(const v8h*)&sx[16 + m][kb], *(const v8h*)&sx[16 + m][kb + 8]);

    dp0 = __builtin_amdgcn_wmma_f32_16x16x32_f16(false, a, false, bp0,
                                                 (short)0, dp0, false, false);
    dp1 = __builtin_amdgcn_wmma_f32_16x16x32_f16(false, a, false, bp1,
                                                 (short)0, dp1, false, false);
    dx0 = __builtin_amdgcn_wmma_f32_16x16x32_f16(false, a, false, bx0,
                                                 (short)0, dx0, false, false);
    dx1 = __builtin_amdgcn_wmma_f32_16x16x32_f16(false, a, false, bx1,
                                                 (short)0, dx1, false, false);
  }

  // Flush per-wave accumulators. D layout: elem j -> row = j + 8*hlf, col = m.
#pragma unroll
  for (int j = 0; j < 8; ++j) {
    const int r = (n * S + (j + 8 * hlf)) * CP;
    atomicAdd(&gsum[r + m],      dp0[j]);
    atomicAdd(&gsum[r + 16 + m], dp1[j]);
    atomicAdd(&gx[r + m],        dx0[j]);
    atomicAdd(&gx[r + 16 + m],   dx1[j]);
  }

  __syncthreads();
  if (tid < S) atomicMax(&glab[tid], labs[tid]);
}

// ---------------------------------------------------------------------------
// Phase 2: finalize (one block, 128 threads: one (n,s) pair each).
// ---------------------------------------------------------------------------
__global__ void scl_phase2(const float* __restrict__ gsum,
                           const float* __restrict__ gx,
                           const int* __restrict__ glab,
                           float* __restrict__ out) {
  using namespace scl;
  __shared__ float s_nll[N * S], s_kl[N * S], s_cnt[N * S], s_pb[S];
  const int tid = threadIdx.x;

  if (tid < N * S) {
    const int ns = tid, s = tid & 15;
    const float* rs = gsum + ns * CP;
    const float* rx = gx + ns * CP;
    const float cnt  = rs[21];          // exact count (WMMA ones column)
    const float corr = rx[21];          // sum of (max + logsumexp)
    const bool  valid = cnt > 0.5f;
    const float safe  = valid ? cnt : 1.0f;

    float mlogm = 0.0f, dot = 0.0f;
#pragma unroll
    for (int c = 0; c < C; ++c) {
      const float mean = valid ? rs[c] / safe : 0.0f;
      const float sl   = rx[c] - corr;  // sum of log p over segment
      if (mean > 0.0f) mlogm += mean * __logf(mean);
      dot += mean * sl;
    }
    s_kl[tid]  = cnt * mlogm - dot;
    s_cnt[tid] = cnt;

    int lab = glab[s];
    lab = (lab >= 0 && lab < C) ? lab : 0;
    const float mlab = valid ? rs[lab] / safe : 0.0f;
    s_nll[tid] = valid ? -__logf(mlab) : 0.0f;
  }
  __syncthreads();

  if (tid < S) {
    float an = 0.0f, ak = 0.0f, ac = 0.0f;
#pragma unroll
    for (int nn = 0; nn < N; ++nn) {
      an += s_nll[nn * S + tid];
      ak += s_kl[nn * S + tid];
      ac += s_cnt[nn * S + tid];
    }
    const float lavg = an / (float)N;
    const float ldev = (ac > 0.0f) ? ak / ((float)C * ac) : 0.0f;
    s_pb[tid] = ALPHA * lavg + BETA * ldev;
  }
  __syncthreads();

  if (tid == 0) {
    float acc = 0.0f;
#pragma unroll
    for (int s = 0; s < S; ++s) acc += s_pb[s];
    out[0] = acc / (float)S;
  }
}

// ---------------------------------------------------------------------------
extern "C" void kernel_launch(void* const* d_in, const int* in_sizes, int n_in,
                              void* d_out, int out_size, void* d_ws,
                              size_t ws_size, hipStream_t stream) {
  using namespace scl;
  (void)in_sizes; (void)n_in; (void)out_size; (void)ws_size;

  const float* logit  = (const float*)d_in[0];
  const int*   blobs  = (const int*)d_in[1];
  const int*   target = (const int*)d_in[2];

  float* gsum = (float*)d_ws;                 // N*S*CP f32
  float* gx   = gsum + N * S * CP;            // N*S*CP f32
  int*   glab = (int*)(gx + N * S * CP);      // S i32

  const int nfloats = 2 * N * S * CP;         // 8192
  scl_zero<<<(nfloats + 255) / 256, 256, 0, stream>>>(gsum, nfloats, glab, S);

  scl_phase1<<<N * BLK_PER_IMG, 256, 0, stream>>>(logit, blobs, target, gsum,
                                                  gx, glab);

  scl_phase2<<<1, 128, 0, stream>>>(gsum, gx, glab, (float*)d_out);
}
